// MultiHeadSelfAttention_8881992368673
// MI455X (gfx1250) — matrix operand order repaired and checked by reading, not compiled, not run on hardware
//
#include <hip/hip_runtime.h>

#define DEV __device__ __forceinline__

typedef __attribute__((ext_vector_type(4)))  float  v4f;
typedef __attribute__((ext_vector_type(8)))  float  v8f;
typedef __attribute__((ext_vector_type(4)))  __bf16 v4bf;
typedef __attribute__((ext_vector_type(8)))  __bf16 v8bf;
typedef __attribute__((ext_vector_type(16))) __bf16 v16bf;

constexpr int D_MODEL   = 1024;
constexpr int NUM_HEADS = 16;
constexpr int HEAD_DIM  = 64;
constexpr int BATCH     = 4;
constexpr int SEQ       = 2048;
constexpr int MTOT      = BATCH * SEQ;   // 8192 rows

// ---------------------------------------------------------------------------
// WMMA helpers (CDNA5 wave32, v_wmma_f32_16x16x32_bf16)
// ---------------------------------------------------------------------------
DEV v8f wmma_bf16(v16bf a, v16bf b, v8f c) {
  // (neg_a, A, neg_b, B, c_mod, C, reuse_a, reuse_b)
  return __builtin_amdgcn_wmma_f32_16x16x32_bf16(false, a, false, b,
                                                 (short)0, c, false, false);
}

// A fragment 16x32 bf16 (M x K). Row-major source, leading dim = ldk elems.
// lane L: row = L%16; elems 0..7 at k = (L>=16?8:0), elems 8..15 at +16.
DEV v16bf load_a_frag(const __bf16* __restrict__ base, int ldk, int lane) {
  const int half = (lane >> 4) & 1;
  const __bf16* p = base + (size_t)(lane & 15) * ldk + half * 8;
  v8bf lo = *(const v8bf*)(p);
  v8bf hi = *(const v8bf*)(p + 16);
  return __builtin_shufflevector(lo, hi, 0,1,2,3,4,5,6,7,8,9,10,11,12,13,14,15);
}

// B fragment 32x16 bf16 (K x N) where B(k,n) = src[n][k] (src row-major, ld=ldk).
// lane L: col n = L%16; elems 0..7 at k = (L>=16?8:0), elems 8..15 at +16.
DEV v16bf load_b_frag(const __bf16* __restrict__ base, int ldk, int lane) {
  const int half = (lane >> 4) & 1;
  const __bf16* p = base + (size_t)(lane & 15) * ldk + half * 8;
  v8bf lo = *(const v8bf*)(p);
  v8bf hi = *(const v8bf*)(p + 16);
  return __builtin_shufflevector(lo, hi, 0,1,2,3,4,5,6,7,8,9,10,11,12,13,14,15);
}

// ---------------------------------------------------------------------------
// fp32 -> bf16 conversion (x and weight matrices)
// ---------------------------------------------------------------------------
__global__ void cvt_bf16_kernel(const float* __restrict__ in,
                                __bf16* __restrict__ out, int n) {
  int i = 4 * (blockIdx.x * blockDim.x + threadIdx.x);
  if (i + 3 < n) {
    v4f f = *(const v4f*)(in + i);
    *(v4bf*)(out + i) = __builtin_convertvector(f, v4bf);
  }
}

// ---------------------------------------------------------------------------
// GEMM:  C[M=8192, N=1024] = A[M,1024](bf16) * W[N,1024](bf16)^T + bias
// mode 0: Q  -> bf16 [B,H,S,Dh], scaled by Dh^-0.5
// mode 1: K  -> bf16 [B,H,S,Dh]
// mode 2: V  -> bf16 [B,H,Dh,S]   (transposed for PV B-fragments)
// mode 3: O  -> f32  [M,N]        (final output to d_out)
// Block = 4 waves; each wave computes a 64x64 tile (4x4 WMMA fragments).
// ---------------------------------------------------------------------------
__global__ __launch_bounds__(128)
void gemm_bf16_kernel(const __bf16* __restrict__ A, const __bf16* __restrict__ W,
                      const float* __restrict__ bias, void* __restrict__ out,
                      int mode, float scale) {
  const int lane = threadIdx.x & 31;
  const int wave = threadIdx.x >> 5;
  const int m0 = blockIdx.x * 256 + wave * 64;
  const int n0 = blockIdx.y * 64;

  v8f acc[4][4] = {};
  for (int k0 = 0; k0 < D_MODEL; k0 += 32) {
    v16bf a[4], b[4];
#pragma unroll
    for (int i = 0; i < 4; ++i)
      a[i] = load_a_frag(A + (size_t)(m0 + 16 * i) * D_MODEL + k0, D_MODEL, lane);
#pragma unroll
    for (int j = 0; j < 4; ++j)
      b[j] = load_b_frag(W + (size_t)(n0 + 16 * j) * D_MODEL + k0, D_MODEL, lane);
#pragma unroll
    for (int i = 0; i < 4; ++i)
#pragma unroll
      for (int j = 0; j < 4; ++j)
        acc[i][j] = wmma_bf16(a[i], b[j], acc[i][j]);
  }

  const int half = lane >> 4;
#pragma unroll
  for (int i = 0; i < 4; ++i) {
#pragma unroll
    for (int j = 0; j < 4; ++j) {
      const int n = n0 + 16 * j + (lane & 15);
      const float bv = bias[n];
#pragma unroll
      for (int r = 0; r < 8; ++r) {
        const int m = m0 + 16 * i + r + 8 * half;
        const float v = (acc[i][j][r] + bv) * scale;
        if (mode == 3) {
          ((float*)out)[(size_t)m * D_MODEL + n] = v;
        } else {
          const int bb = m >> 11, s = m & (SEQ - 1);
          const int h = n >> 6, dh = n & 63;
          size_t idx;
          if (mode == 2)
            idx = ((size_t)(bb * NUM_HEADS + h) * HEAD_DIM + dh) * SEQ + s;
          else
            idx = ((size_t)(bb * NUM_HEADS + h) * SEQ + s) * HEAD_DIM + dh;
          ((__bf16*)out)[idx] = (__bf16)v;
        }
      }
    }
  }
}

// ---------------------------------------------------------------------------
// Flash attention: 1 wave = 32 query rows of one (b,h); streams keys 64/iter.
// Q pre-scaled by Dh^-0.5.  P routed C-layout -> LDS -> A-fragment layout.
// ---------------------------------------------------------------------------
__global__ __launch_bounds__(128)
void flash_attn_kernel(const __bf16* __restrict__ Q, const __bf16* __restrict__ K,
                       const __bf16* __restrict__ Vt, __bf16* __restrict__ ctx) {
  __shared__ __attribute__((aligned(16))) __bf16 Pbuf[4][32][72];

  const int lane = threadIdx.x & 31;
  const int wave = threadIdx.x >> 5;
  const int half = lane >> 4;
  const int bh = blockIdx.y;             // b*H + h
  const int b = bh >> 4, h = bh & 15;
  const int q0 = blockIdx.x * 128 + wave * 32;

  const __bf16* Qb = Q + (size_t)bh * SEQ * HEAD_DIM;
  const __bf16* Kb = K + (size_t)bh * SEQ * HEAD_DIM;
  const __bf16* Vb = Vt + (size_t)bh * HEAD_DIM * SEQ;

  v16bf qf[2][2];
#pragma unroll
  for (int mi = 0; mi < 2; ++mi)
#pragma unroll
    for (int kk = 0; kk < 2; ++kk)
      qf[mi][kk] = load_a_frag(Qb + (size_t)(q0 + 16 * mi) * HEAD_DIM + kk * 32,
                               HEAD_DIM, lane);

  v8f o[2][4] = {};
  float mrow[2][8], lrow[2][8];
#pragma unroll
  for (int mi = 0; mi < 2; ++mi)
#pragma unroll
    for (int r = 0; r < 8; ++r) { mrow[mi][r] = -1e30f; lrow[mi][r] = 0.f; }

  for (int kv = 0; kv < SEQ; kv += 64) {
    // ---- S = Q * K^T (32x64 tile) ----
    v8f s[2][4] = {};
#pragma unroll
    for (int ni = 0; ni < 4; ++ni) {
#pragma unroll
      for (int kk = 0; kk < 2; ++kk) {
        v16bf kf = load_b_frag(Kb + (size_t)(kv + 16 * ni) * HEAD_DIM + kk * 32,
                               HEAD_DIM, lane);
#pragma unroll
        for (int mi = 0; mi < 2; ++mi)
          s[mi][ni] = wmma_bf16(qf[mi][kk], kf, s[mi][ni]);
      }
    }
    // ---- online softmax (rows live in 16-lane groups) ----
#pragma unroll
    for (int mi = 0; mi < 2; ++mi) {
#pragma unroll
      for (int r = 0; r < 8; ++r) {
        float mx = fmaxf(fmaxf(s[mi][0][r], s[mi][1][r]),
                         fmaxf(s[mi][2][r], s[mi][3][r]));
#pragma unroll
        for (int msk = 1; msk < 16; msk <<= 1)
          mx = fmaxf(mx, __shfl_xor(mx, msk, 32));
        const float mnew = fmaxf(mrow[mi][r], mx);
        const float alpha = __expf(mrow[mi][r] - mnew);
        mrow[mi][r] = mnew;
        float rs = 0.f;
#pragma unroll
        for (int ni = 0; ni < 4; ++ni) {
          const float p = __expf(s[mi][ni][r] - mnew);
          s[mi][ni][r] = p;
          rs += p;
        }
#pragma unroll
        for (int msk = 1; msk < 16; msk <<= 1)
          rs += __shfl_xor(rs, msk, 32);
        lrow[mi][r] = lrow[mi][r] * alpha + rs;
#pragma unroll
        for (int nd = 0; nd < 4; ++nd)
          o[mi][nd][r] *= alpha;
      }
    }
    // ---- P (C-layout) -> LDS -> A-fragment layout ----
#pragma unroll
    for (int mi = 0; mi < 2; ++mi)
#pragma unroll
      for (int ni = 0; ni < 4; ++ni)
#pragma unroll
        for (int r = 0; r < 8; ++r)
          Pbuf[wave][16 * mi + r + 8 * half][16 * ni + (lane & 15)] =
              (__bf16)s[mi][ni][r];
    asm volatile("s_wait_dscnt 0x0" ::: "memory");
    v16bf pf[2][2];
#pragma unroll
    for (int mi = 0; mi < 2; ++mi)
#pragma unroll
      for (int kk = 0; kk < 2; ++kk) {
        const __bf16* p = &Pbuf[wave][16 * mi + (lane & 15)][kk * 32 + half * 8];
        v8bf lo = *(const v8bf*)(p);
        v8bf hi = *(const v8bf*)(p + 16);
        pf[mi][kk] = __builtin_shufflevector(lo, hi,
                       0,1,2,3,4,5,6,7,8,9,10,11,12,13,14,15);
      }
    // ---- O += P * V  (V transposed -> contiguous B-fragment loads) ----
#pragma unroll
    for (int kk = 0; kk < 2; ++kk)
#pragma unroll
      for (int nd = 0; nd < 4; ++nd) {
        v16bf vf = load_b_frag(Vb + (size_t)(16 * nd) * SEQ + kv + kk * 32,
                               SEQ, lane);
#pragma unroll
        for (int mi = 0; mi < 2; ++mi)
          o[mi][nd] = wmma_bf16(pf[mi][kk], vf, o[mi][nd]);
      }
  }
  // ---- normalize and store ctx [B,S,H,Dh] (bf16) ----
#pragma unroll
  for (int mi = 0; mi < 2; ++mi)
#pragma unroll
    for (int nd = 0; nd < 4; ++nd) {
      const int dh = 16 * nd + (lane & 15);
#pragma unroll
      for (int r = 0; r < 8; ++r) {
        const int srow = q0 + 16 * mi + r + 8 * half;
        const float v = o[mi][nd][r] / lrow[mi][r];
        ctx[((size_t)(b * SEQ + srow) * NUM_HEADS + h) * HEAD_DIM + dh] = (__bf16)v;
      }
    }
}

// ---------------------------------------------------------------------------
extern "C" void kernel_launch(void* const* d_in, const int* in_sizes, int n_in,
                              void* d_out, int out_size, void* d_ws, size_t ws_size,
                              hipStream_t stream) {
  (void)in_sizes; (void)n_in; (void)out_size; (void)ws_size;
  const float* x  = (const float*)d_in[0];
  const float* Wq = (const float*)d_in[1];
  const float* bq = (const float*)d_in[2];
  const float* Wk = (const float*)d_in[3];
  const float* bk = (const float*)d_in[4];
  const float* Wv = (const float*)d_in[5];
  const float* bv = (const float*)d_in[6];
  const float* Wo = (const float*)d_in[7];
  const float* bo = (const float*)d_in[8];

  char* ws = (char*)d_ws;
  const size_t MB = 1ull << 20;
  __bf16* xb  = (__bf16*)(ws + 0);        // 16 MB  x in bf16
  __bf16* wqb = (__bf16*)(ws + 16 * MB);  //  2 MB
  __bf16* wkb = (__bf16*)(ws + 18 * MB);  //  2 MB
  __bf16* wvb = (__bf16*)(ws + 20 * MB);  //  2 MB
  __bf16* wob = (__bf16*)(ws + 22 * MB);  //  2 MB
  __bf16* Qh  = (__bf16*)(ws + 24 * MB);  // 16 MB  [B,H,S,Dh] pre-scaled
  __bf16* Kh  = (__bf16*)(ws + 40 * MB);  // 16 MB  [B,H,S,Dh]
  __bf16* Vt  = (__bf16*)(ws + 56 * MB);  // 16 MB  [B,H,Dh,S]
  __bf16* ctx = (__bf16*)(ws + 72 * MB);  // 16 MB  [B,S,H,Dh]

  const int nx = MTOT * D_MODEL;          // 8,388,608
  const int nw = D_MODEL * D_MODEL;       // 1,048,576
  cvt_bf16_kernel<<<dim3((nx / 4 + 255) / 256), dim3(256), 0, stream>>>(x, xb, nx);
  cvt_bf16_kernel<<<dim3((nw / 4 + 255) / 256), dim3(256), 0, stream>>>(Wq, wqb, nw);
  cvt_bf16_kernel<<<dim3((nw / 4 + 255) / 256), dim3(256), 0, stream>>>(Wk, wkb, nw);
  cvt_bf16_kernel<<<dim3((nw / 4 + 255) / 256), dim3(256), 0, stream>>>(Wv, wvb, nw);
  cvt_bf16_kernel<<<dim3((nw / 4 + 255) / 256), dim3(256), 0, stream>>>(Wo, wob, nw);

  const dim3 ggrid(MTOT / 256, D_MODEL / 64);
  const dim3 gblk(128);
  const float qscale = 0.125f;            // HEAD_DIM^-0.5
  gemm_bf16_kernel<<<ggrid, gblk, 0, stream>>>(xb, wqb, bq, Qh, 0, qscale);
  gemm_bf16_kernel<<<ggrid, gblk, 0, stream>>>(xb, wkb, bk, Kh, 1, 1.0f);
  gemm_bf16_kernel<<<ggrid, gblk, 0, stream>>>(xb, wvb, bv, Vt, 2, 1.0f);

  flash_attn_kernel<<<dim3(SEQ / 128, BATCH * NUM_HEADS), dim3(128), 0, stream>>>(
      Qh, Kh, Vt, ctx);

  gemm_bf16_kernel<<<ggrid, gblk, 0, stream>>>(ctx, wob, bo, d_out, 3, 1.0f);
}
